// SSGAT_37752762532112
// MI455X (gfx1250) — matrix operand order repaired and checked by reading, not compiled, not run on hardware
//
#include <hip/hip_runtime.h>
#include <hip/hip_bf16.h>
#include <hip/hip_fp16.h>

#define NNODES 100000
#define NEDGES 1600000
#define INF_   128
#define HID    32
#define HC     64          // H * C = 2 * 32
#define EPS_   1e-5f
#define NEG_SLOPE 0.2f

typedef __attribute__((ext_vector_type(16))) _Float16 v16h;
typedef __attribute__((ext_vector_type(8)))  _Float16 v8h;
typedef __attribute__((ext_vector_type(4)))  _Float16 v4h;
typedef __attribute__((ext_vector_type(2)))  _Float16 v2h;
typedef __attribute__((ext_vector_type(8)))  float    v8f;

// ---------------- WMMA fragment loaders (CDNA5 wave32 layouts) ----------------
// A: 16x32 f16 tile, row-major in LDS (row stride ld halfs).
// Lane L: row M=L&15, K split at lane 16.  a[0..7] = K 16g..16g+7 (contiguous),
// a[8..15] = K 16g+8..16g+15 (contiguous) -> two ds_load_b128.
__device__ inline v16h frag_a(const _Float16* p, int ld) {
  const int lane = threadIdx.x & 31;
  const int row  = lane & 15;
  const int grp  = lane >> 4;
  const _Float16* b0 = p + row * ld + grp * 16;
  v8h lo = *(const v8h*)(b0);
  v8h hi = *(const v8h*)(b0 + 8);
  return __builtin_shufflevector(lo, hi, 0,1,2,3,4,5,6,7,8,9,10,11,12,13,14,15);
}

// B: stored PRE-SWIZZLED in LDS in fragment-linear order:
//   buf[ (tile*32 + lane)*16 + t ]  holds element (k = kt*32 + (lane>>4)*16 + t,
//                                                  col = jt*16 + (lane&15))
// so a fragment is 32 contiguous bytes per lane -> two ds_load_b128.
__device__ inline v16h frag_bs(const _Float16* buf, int tile) {
  const _Float16* p = buf + ((tile << 5) + (threadIdx.x & 31)) * 16;
  v8h lo = *(const v8h*)(p);
  v8h hi = *(const v8h*)(p + 8);
  return __builtin_shufflevector(lo, hi, 0,1,2,3,4,5,6,7,8,9,10,11,12,13,14,15);
}

// cooperative fill of a swizzled B buffer from a row-major [Ktot x N] f32 weight
__device__ inline void fill_swz(_Float16* dst, const float* __restrict__ W,
                                int Ktot, int N) {
  const int NJ = N >> 4;
  for (int i = threadIdx.x; i < Ktot * N; i += (int)blockDim.x) {
    int t    = i & 15;
    int L    = (i >> 4) & 31;
    int tile = i >> 9;
    int jt   = tile % NJ;
    int kt   = tile / NJ;
    int k    = kt * 32 + ((L >> 4) << 4) + t;
    int c    = (jt << 4) + (L & 15);
    dst[i] = (_Float16)W[k * N + c];
  }
}

#define WMMA_F16(A, B, C) \
  __builtin_amdgcn_wmma_f32_16x16x32_f16(false, (A), false, (B), (short)0, (C), false, false)

// ---------------- ordered-uint key for float atomicMax ----------------
__device__ inline unsigned fkey(float f) {
  unsigned u = __float_as_uint(f);
  return (u & 0x80000000u) ? ~u : (u | 0x80000000u);
}
__device__ inline float fdec(unsigned k) {
  unsigned u = (k & 0x80000000u) ? (k & 0x7FFFFFFFu) : ~k;
  return __uint_as_float(u);
}

// =====================================================================
// Kernel 1: h = relu(BN1(x@W1)); h = relu(BN2(h@W2)); hw = h@Wg;
//           a_src/a_dst = <hw, att>.  One wave per 16-node tile; WMMA GEMMs.
// =====================================================================
__global__ __launch_bounds__(128) void k_node_front(
    const float* __restrict__ x,
    const float* __restrict__ W1, const float* __restrict__ b1,
    const float* __restrict__ g1, const float* __restrict__ beta1,
    const float* __restrict__ rm1, const float* __restrict__ rv1,
    const float* __restrict__ W2, const float* __restrict__ b2,
    const float* __restrict__ g2, const float* __restrict__ beta2,
    const float* __restrict__ rm2, const float* __restrict__ rv2,
    const float* __restrict__ Wg,
    const float* __restrict__ att_src, const float* __restrict__ att_dst,
    float* __restrict__ hw, float* __restrict__ asrc, float* __restrict__ adst)
{
  __shared__ __align__(16) _Float16 w1s[INF_ * HID];   // swizzled 128x32
  __shared__ __align__(16) _Float16 w2s[HID * HID];    // swizzled 32x32
  __shared__ __align__(16) _Float16 wgs[HID * HC];     // swizzled 32x64
  __shared__ __align__(16) _Float16 xh [4][16 * INF_];
  __shared__ __align__(16) _Float16 t1h[4][16 * HID];
  __shared__ __align__(16) _Float16 t2h[4][16 * HID];
  __shared__ __align__(16) float    hwf[4][16 * HC];

  fill_swz(w1s, W1, INF_, HID);
  fill_swz(w2s, W2, HID, HID);
  fill_swz(wgs, Wg, HID, HC);

  const int tid  = threadIdx.x;
  const int wave = tid >> 5;
  const int lane = tid & 31;
  const int col  = lane & 15;
  const int grp  = lane >> 4;
  const int base = blockIdx.x * 64 + wave * 16;

  // stage x tile (f32 -> f16), float4 loads + b64 LDS stores
  for (int i = lane; i < 16 * INF_ / 4; i += 32) {
    int r  = i >> 5;               // 32 quads per row
    int k4 = (i & 31) * 4;
    int n  = base + r;
    v4h hv;
    if (n < NNODES) {
      const float4 f = *(const float4*)(x + (size_t)n * INF_ + k4);
      hv[0] = (_Float16)f.x; hv[1] = (_Float16)f.y;
      hv[2] = (_Float16)f.z; hv[3] = (_Float16)f.w;
    } else {
      hv[0] = hv[1] = hv[2] = hv[3] = (_Float16)0.f;
    }
    *(v4h*)&xh[wave][r * INF_ + k4] = hv;
  }
  __syncthreads();

  // ---- layer 1: [16x128] @ [128x32], K in 4 chunks of 32 ----
#pragma unroll
  for (int jt = 0; jt < 2; ++jt) {
    v8f acc = {};
#pragma unroll
    for (int kt = 0; kt < 4; ++kt) {
      v16h a = frag_a(&xh[wave][kt * 32], INF_);
      v16h b = frag_bs(w1s, kt * 2 + jt);
      acc = WMMA_F16(a, b, acc);
    }
    int j = jt * 16 + col;
    float sc = g1[j] * rsqrtf(rv1[j] + EPS_);
    float sh = beta1[j] + sc * (b1[j] - rm1[j]);
#pragma unroll
    for (int r = 0; r < 8; ++r)
      t1h[wave][(r + 8 * grp) * HID + j] = (_Float16)fmaxf(acc[r] * sc + sh, 0.f);
  }
  __syncthreads();

  // ---- layer 2: [16x32] @ [32x32] ----
#pragma unroll
  for (int jt = 0; jt < 2; ++jt) {
    v16h a = frag_a(&t1h[wave][0], HID);
    v16h b = frag_bs(w2s, jt);
    v8f acc = {};
    acc = WMMA_F16(a, b, acc);
    int j = jt * 16 + col;
    float sc = g2[j] * rsqrtf(rv2[j] + EPS_);
    float sh = beta2[j] + sc * (b2[j] - rm2[j]);
#pragma unroll
    for (int r = 0; r < 8; ++r)
      t2h[wave][(r + 8 * grp) * HID + j] = (_Float16)fmaxf(acc[r] * sc + sh, 0.f);
  }
  __syncthreads();

  // ---- GAT projection: [16x32] @ [32x64] (bg added post-aggregation) ----
#pragma unroll
  for (int jt = 0; jt < 4; ++jt) {
    v16h a = frag_a(&t2h[wave][0], HID);
    v16h b = frag_bs(wgs, jt);
    v8f acc = {};
    acc = WMMA_F16(a, b, acc);
    int j = jt * 16 + col;
#pragma unroll
    for (int r = 0; r < 8; ++r)
      hwf[wave][(r + 8 * grp) * HC + j] = acc[r];
  }
  __syncthreads();

  // coalesced hw store (float4) + attention logits
  for (int i = lane; i < 16 * HC / 4; i += 32) {
    int r  = i >> 4;               // 16 quads per row
    int c4 = (i & 15) * 4;
    int n  = base + r;
    if (n < NNODES) {
      const float4 v = *(const float4*)&hwf[wave][r * HC + c4];
      *(float4*)(hw + (size_t)n * HC + c4) = v;
    }
  }
  if (lane < 16) {
    int n = base + lane;
    if (n < NNODES) {
#pragma unroll
      for (int h = 0; h < 2; ++h) {
        float s0 = 0.f, s1 = 0.f;
        for (int c = 0; c < 32; ++c) {
          float v = hwf[wave][lane * HC + h * 32 + c];
          s0 += v * att_src[h * 32 + c];
          s1 += v * att_dst[h * 32 + c];
        }
        asrc[n * 2 + h] = s0;
        adst[n * 2 + h] = s1;
      }
    }
  }
}

// =====================================================================
// Kernel 2: workspace init (segment-max keys, sums, aggregation buffer)
// =====================================================================
__global__ void k_init(float* __restrict__ agg, unsigned* __restrict__ mkey,
                       float* __restrict__ ssum)
{
  int i = blockIdx.x * blockDim.x + threadIdx.x;
  if (i < NNODES * HC) agg[i] = 0.f;
  if (i < NNODES * 2) { mkey[i] = 0u; ssum[i] = 0.f; }   // key 0 == -inf
}

// =====================================================================
// Kernel 3: segment max of leaky_relu(a_src[src] + a_dst[dst]) over dst
// =====================================================================
__global__ void k_segmax(const long long* __restrict__ ei,
                         const float* __restrict__ asrc, const float* __restrict__ adst,
                         unsigned* __restrict__ mkey)
{
  int e = blockIdx.x * blockDim.x + threadIdx.x;
  if (e >= NEDGES + NNODES) return;
  int s, d;
  if (e < NEDGES) { s = (int)ei[e]; d = (int)ei[NEDGES + e]; }
  else            { s = d = e - NEDGES; }
#pragma unroll
  for (int h = 0; h < 2; ++h) {
    float a = asrc[s * 2 + h] + adst[d * 2 + h];
    a = (a > 0.f) ? a : a * NEG_SLOPE;
    atomicMax(&mkey[d * 2 + h], fkey(a));
  }
}

// =====================================================================
// Kernel 4: segment sum of exp(a - m[dst]) over dst
// =====================================================================
__global__ void k_segsum(const long long* __restrict__ ei,
                         const float* __restrict__ asrc, const float* __restrict__ adst,
                         const unsigned* __restrict__ mkey, float* __restrict__ ssum)
{
  int e = blockIdx.x * blockDim.x + threadIdx.x;
  if (e >= NEDGES + NNODES) return;
  int s, d;
  if (e < NEDGES) { s = (int)ei[e]; d = (int)ei[NEDGES + e]; }
  else            { s = d = e - NEDGES; }
#pragma unroll
  for (int h = 0; h < 2; ++h) {
    float a = asrc[s * 2 + h] + adst[d * 2 + h];
    a = (a > 0.f) ? a : a * NEG_SLOPE;
    float m = fdec(mkey[d * 2 + h]);
    atomicAdd(&ssum[d * 2 + h], __expf(a - m));
  }
}

// =====================================================================
// Kernel 5: message scatter  agg[dst] += hw[src] * alpha   (float4 lanes)
// =====================================================================
__global__ void k_message(const long long* __restrict__ ei,
                          const float* __restrict__ asrc, const float* __restrict__ adst,
                          const unsigned* __restrict__ mkey, const float* __restrict__ ssum,
                          const float* __restrict__ hw, float* __restrict__ agg)
{
  int gid = blockIdx.x * blockDim.x + threadIdx.x;
  if (gid >= (NEDGES + NNODES) * 16) return;
  int e  = gid >> 4;
  int c4 = (gid & 15) * 4;
  int h  = c4 >> 5;
  int s, d;
  if (e < NEDGES) { s = (int)ei[e]; d = (int)ei[NEDGES + e]; }
  else            { s = d = e - NEDGES; }
  float a = asrc[s * 2 + h] + adst[d * 2 + h];
  a = (a > 0.f) ? a : a * NEG_SLOPE;
  float m = fdec(mkey[d * 2 + h]);
  float alpha = __expf(a - m) / (ssum[d * 2 + h] + 1e-16f);
  const float4 hv = *(const float4*)(hw + (size_t)s * HC + c4);
  float* o = agg + (size_t)d * HC + c4;
  atomicAdd(o + 0, hv.x * alpha);
  atomicAdd(o + 1, hv.y * alpha);
  atomicAdd(o + 2, hv.z * alpha);
  atomicAdd(o + 3, hv.w * alpha);
}

// =====================================================================
// Kernel 6: y = elu(LN(agg + bg)); h = relu(BN3(y@W3)); h = relu(BN4(h@W4));
//           out = h@W5 + b5.   WMMA for the 64->32 and 32->16 GEMMs.
// =====================================================================
__global__ __launch_bounds__(128) void k_node_back(
    const float* __restrict__ agg, const float* __restrict__ bg,
    const float* __restrict__ lng, const float* __restrict__ lnb,
    const float* __restrict__ W3, const float* __restrict__ b3,
    const float* __restrict__ g3, const float* __restrict__ beta3,
    const float* __restrict__ rm3, const float* __restrict__ rv3,
    const float* __restrict__ W4, const float* __restrict__ b4,
    const float* __restrict__ g4, const float* __restrict__ beta4,
    const float* __restrict__ rm4, const float* __restrict__ rv4,
    const float* __restrict__ W5, const float* __restrict__ b5,
    float* __restrict__ out)
{
  __shared__ __align__(16) _Float16 w3s[HC * HID];     // swizzled 64x32
  __shared__ __align__(16) _Float16 w4s[HID * 16];     // swizzled 32x16
  __shared__ __align__(16) float    yst[4][16 * HC];
  __shared__ __align__(16) _Float16 yh [4][16 * HC];
  __shared__ __align__(16) _Float16 t3h[4][16 * HID];
  __shared__ __align__(16) float    t4f[4][16 * 16];

  fill_swz(w3s, W3, HC, HID);
  fill_swz(w4s, W4, HID, 16);

  const int tid  = threadIdx.x;
  const int wave = tid >> 5;
  const int lane = tid & 31;
  const int col  = lane & 15;
  const int grp  = lane >> 4;
  const int base = blockIdx.x * 64 + wave * 16;

  for (int i = lane; i < 16 * HC / 4; i += 32) {
    int r  = i >> 4;
    int c4 = (i & 15) * 4;
    int n  = base + r;
    float4 v;
    if (n < NNODES) {
      const float4 av = *(const float4*)(agg + (size_t)n * HC + c4);
      const float4 bv = *(const float4*)(bg + c4);
      v.x = av.x + bv.x; v.y = av.y + bv.y;
      v.z = av.z + bv.z; v.w = av.w + bv.w;
    } else {
      v.x = v.y = v.z = v.w = 0.f;
    }
    *(float4*)&yst[wave][r * HC + c4] = v;
  }
  __syncthreads();

  // LayerNorm + ELU per row, emit f16 A tile (paired b32 stores)
  if (lane < 16) {
    float mu = 0.f;
    for (int c = 0; c < HC; ++c) mu += yst[wave][lane * HC + c];
    mu *= (1.f / HC);
    float var = 0.f;
    for (int c = 0; c < HC; ++c) {
      float dlt = yst[wave][lane * HC + c] - mu;
      var += dlt * dlt;
    }
    var *= (1.f / HC);
    float rs = rsqrtf(var + EPS_);
    for (int c = 0; c < HC; c += 2) {
      float v0 = lng[c]     * (yst[wave][lane * HC + c]     - mu) * rs + lnb[c];
      float v1 = lng[c + 1] * (yst[wave][lane * HC + c + 1] - mu) * rs + lnb[c + 1];
      v0 = (v0 > 0.f) ? v0 : (__expf(v0) - 1.f);   // ELU
      v1 = (v1 > 0.f) ? v1 : (__expf(v1) - 1.f);
      v2h pk; pk[0] = (_Float16)v0; pk[1] = (_Float16)v1;
      *(v2h*)&yh[wave][lane * HC + c] = pk;
    }
  }
  __syncthreads();

  // ---- [16x64] @ [64x32], K in 2 chunks ----
#pragma unroll
  for (int jt = 0; jt < 2; ++jt) {
    v8f acc = {};
#pragma unroll
    for (int kt = 0; kt < 2; ++kt) {
      v16h a = frag_a(&yh[wave][kt * 32], HC);
      v16h b = frag_bs(w3s, kt * 2 + jt);
      acc = WMMA_F16(a, b, acc);
    }
    int j = jt * 16 + col;
    float sc = g3[j] * rsqrtf(rv3[j] + EPS_);
    float sh = beta3[j] + sc * (b3[j] - rm3[j]);
#pragma unroll
    for (int r = 0; r < 8; ++r)
      t3h[wave][(r + 8 * grp) * HID + j] = (_Float16)fmaxf(acc[r] * sc + sh, 0.f);
  }
  __syncthreads();

  // ---- [16x32] @ [32x16] ----
  {
    v16h a = frag_a(&t3h[wave][0], HID);
    v16h b = frag_bs(w4s, 0);
    v8f acc = {};
    acc = WMMA_F16(a, b, acc);
    int j = col;
    float sc = g4[j] * rsqrtf(rv4[j] + EPS_);
    float sh = beta4[j] + sc * (b4[j] - rm4[j]);
#pragma unroll
    for (int r = 0; r < 8; ++r)
      t4f[wave][(r + 8 * grp) * 16 + j] = fmaxf(acc[r] * sc + sh, 0.f);
  }
  __syncthreads();

  // ---- [16x16] @ [16x1] + b5 ----
  if (lane < 16) {
    int n = base + lane;
    if (n < NNODES) {
      float acc = b5[0];
      for (int j = 0; j < 16; ++j) acc += t4f[wave][lane * 16 + j] * W5[j];
      out[n] = acc;
    }
  }
}

// =====================================================================
extern "C" void kernel_launch(void* const* d_in, const int* in_sizes, int n_in,
                              void* d_out, int out_size, void* d_ws, size_t ws_size,
                              hipStream_t stream) {
  const float* x      = (const float*)d_in[0];
  const long long* ei = (const long long*)d_in[1];   // int64 [2, E]
  const float* W1   = (const float*)d_in[2];
  const float* b1   = (const float*)d_in[3];
  const float* g1   = (const float*)d_in[4];
  const float* beta1= (const float*)d_in[5];
  const float* rm1  = (const float*)d_in[6];
  const float* rv1  = (const float*)d_in[7];
  const float* W2   = (const float*)d_in[8];
  const float* b2   = (const float*)d_in[9];
  const float* g2   = (const float*)d_in[10];
  const float* beta2= (const float*)d_in[11];
  const float* rm2  = (const float*)d_in[12];
  const float* rv2  = (const float*)d_in[13];
  const float* Wg   = (const float*)d_in[14];
  const float* att_src = (const float*)d_in[15];
  const float* att_dst = (const float*)d_in[16];
  const float* bg   = (const float*)d_in[17];
  const float* lng  = (const float*)d_in[18];
  const float* lnb  = (const float*)d_in[19];
  const float* W3   = (const float*)d_in[20];
  const float* b3   = (const float*)d_in[21];
  const float* g3   = (const float*)d_in[22];
  const float* beta3= (const float*)d_in[23];
  const float* rm3  = (const float*)d_in[24];
  const float* rv3  = (const float*)d_in[25];
  const float* W4   = (const float*)d_in[26];
  const float* b4   = (const float*)d_in[27];
  const float* g4   = (const float*)d_in[28];
  const float* beta4= (const float*)d_in[29];
  const float* rm4  = (const float*)d_in[30];
  const float* rv4  = (const float*)d_in[31];
  const float* W5   = (const float*)d_in[32];
  const float* b5   = (const float*)d_in[33];
  float* out = (float*)d_out;

  // workspace layout (all 4-byte types)
  float*    hw   = (float*)d_ws;                          // N * 64
  float*    asrc = hw   + (size_t)NNODES * HC;            // N * 2
  float*    adst = asrc + (size_t)NNODES * 2;             // N * 2
  unsigned* mkey = (unsigned*)(adst + (size_t)NNODES * 2);// N * 2
  float*    ssum = (float*)mkey + (size_t)NNODES * 2;     // N * 2
  float*    agg  = ssum + (size_t)NNODES * 2;             // N * 64

  const int nodeBlocks = (NNODES + 63) / 64;

  k_node_front<<<nodeBlocks, 128, 0, stream>>>(
      x, W1, b1, g1, beta1, rm1, rv1,
      W2, b2, g2, beta2, rm2, rv2,
      Wg, att_src, att_dst, hw, asrc, adst);

  k_init<<<(NNODES * HC + 255) / 256, 256, 0, stream>>>(agg, mkey, ssum);

  const int nE = NEDGES + NNODES;
  k_segmax<<<(nE + 255) / 256, 256, 0, stream>>>(ei, asrc, adst, mkey);
  k_segsum<<<(nE + 255) / 256, 256, 0, stream>>>(ei, asrc, adst, mkey, ssum);
  k_message<<<(nE * 16 + 255) / 256, 256, 0, stream>>>(ei, asrc, adst, mkey, ssum, hw, agg);

  k_node_back<<<nodeBlocks, 128, 0, stream>>>(
      agg, bg, lng, lnb,
      W3, b3, g3, beta3, rm3, rv3,
      W4, b4, g4, beta4, rm4, rv4,
      W5, b5, out);
}
